// attention_62706522521712
// MI455X (gfx1250) — matrix operand order repaired and checked by reading, not compiled, not run on hardware
//
#include <hip/hip_runtime.h>
#include <hip/hip_bf16.h>

typedef __bf16 bf16;
typedef __attribute__((ext_vector_type(16))) __bf16 v16bf;
typedef __attribute__((ext_vector_type(8)))  float  v8f;
typedef __attribute__((ext_vector_type(4)))  unsigned int u32x4;
typedef __attribute__((ext_vector_type(8)))  int i32x8;
typedef __attribute__((ext_vector_type(4)))  int i32x4;
typedef unsigned long long ull;

#if defined(__gfx1250__) && __has_builtin(__builtin_amdgcn_tensor_load_to_lds)
#define USE_TDM 1
#else
#define USE_TDM 0
#endif

// ---------------------------------------------------------------------------
// fp32 -> bf16 conversion (one-time, precedes the GEMM pipeline)
// ---------------------------------------------------------------------------
__global__ __launch_bounds__(256) void f32_to_bf16_kernel(const float* __restrict__ x,
                                                          bf16* __restrict__ y, int n) {
  int i = blockIdx.x * 256 + threadIdx.x;
  if (i < n) y[i] = (bf16)x[i];
}

#if USE_TDM
// ---------------------------------------------------------------------------
// TDM: DMA one 128x32 bf16 A-tile (row-major, row stride = lda elements) from
// global memory into LDS. The TDM pad feature inserts 4 DWORDs after every 16
// DWORDs stored (= one 64B tile row), giving the padded 40-element LDS row
// stride the WMMA fragment reads expect (conflict-free banking).
// Descriptor per CDNA5 ISA ch.8 (D# groups 0/1; 2D tile -> groups 2/3 zero).
// This toolchain exposes the 6-arg builtin form (extra descriptor group).
// ---------------------------------------------------------------------------
__device__ __forceinline__ void tdm_load_tile_a(const bf16* gptr, unsigned lds_off,
                                                int lda) {
  const ull ga = (ull)gptr;
  u32x4 g0;
  g0[0] = 1u;                                                  // count=1 (valid)
  g0[1] = lds_off;                                             // LDS byte address
  g0[2] = (unsigned)ga;                                        // global_addr[31:0]
  g0[3] = (unsigned)((ga >> 32) & 0x01FFFFFFu) | 0x80000000u;  // addr[56:32], type=2
  const unsigned dim0 = 0x40000000u;  // huge tensor dims: tiles are in-bounds,
  const unsigned dim1 = 0x40000000u;  // never trigger OOB zero-fill
  i32x8 g1;
  g1[0] = (int)((1u << 16) |           // data_size = 2 bytes
                (1u << 20) |           // pad_enable
                (3u << 22) |           // pad_interval: 16 DWORDs (one 64B row)
                (3u << 25));           // pad_amount:   4 DWORDs (8 elements)
  g1[1] = (int)((dim0 & 0xFFFFu) << 16);                       // tensor_dim0 lo16
  g1[2] = (int)((dim0 >> 16) | ((dim1 & 0xFFFFu) << 16));      // dim0 hi / dim1 lo
  g1[3] = (int)((dim1 >> 16) | (32u << 16));                   // dim1 hi / tile_dim0=32
  g1[4] = 128;                                                 // tile_dim1=128, tile_dim2=0
  g1[5] = lda;                                                 // tensor_dim0_stride lo32
  g1[6] = 0;                                                   // stride hi / dim1_stride
  g1[7] = 0;
  const i32x4 gz4 = {0, 0, 0, 0};
  const i32x8 gz8 = {0, 0, 0, 0, 0, 0, 0, 0};
  __builtin_amdgcn_tensor_load_to_lds(g0, g1, gz4, gz4, gz8, 0);
}
#endif

// ---------------------------------------------------------------------------
// Tiled bf16 GEMM with fp32 accumulation on v_wmma_f32_16x16x32_bf16.
//   C[z] = alpha * A[z] @ B[z] (+ bias[z1])
// Block tile 128x128, K-tile 32, LDS double-buffered. 8 waves (wave32):
// 2 along M x 4 along N, each wave computes a 64x32 tile as 4x2 fragments.
// A tiles arrive via the Tensor Data Mover (wave 0 issues, TENSORcnt sync);
// B tiles are staged transposed ([n][k]) so each lane's 16 K-elements are
// two 16B LDS reads (k = kg*8.. and 16+kg*8..) matching the ISA B-fragment layout.
//
// OUT_MODE: 0 = fp32 output, 1 = bf16 output,
//           2 = bf16 transposed output:
//               C[row_hi*strideRB + col*ldtrans + row_lo], row split at 1024.
// Batch z = blockIdx.z, z1 = z / Z2, z2 = z % Z2; offsets z1*s?1 + z2*s?2.
// ---------------------------------------------------------------------------
template <int OUT_MODE, bool HAS_BIAS>
__global__ __launch_bounds__(256) void gemm_bf16_wmma(
    const bf16* __restrict__ A, const bf16* __restrict__ Bm,
    void* __restrict__ Cv, const float* __restrict__ bias,
    int K, int lda, int ldb, int ldc,
    int Z2,
    ull sA1, ull sA2, ull sB1, ull sB2, ull sC1, ull sC2, ull sBias1,
    ull strideRB, int ldtrans,
    float alpha) {
  __shared__ bf16 sA[2][128 * 40];
  __shared__ bf16 sB[2][128 * 40];

  const int z = blockIdx.z;
  const int z1 = z / Z2;
  const int z2 = z % Z2;
  A += (ull)z1 * sA1 + (ull)z2 * sA2;
  Bm += (ull)z1 * sB1 + (ull)z2 * sB2;
  const ull coff = (ull)z1 * sC1 + (ull)z2 * sC2;
  if (HAS_BIAS) bias += (ull)z1 * sBias1;

  const int tid = threadIdx.x;
  const int lane = tid & 31;
  const int l16 = lane & 15;   // row/col within 16x16 fragment
  const int kg = lane >> 4;    // K-group (lane half) per ISA fragment layout
  const int wid = tid >> 5;
  const int wm = wid & 1;      // 2 waves along M (64 rows each)
  const int wn = wid >> 1;     // 4 waves along N (32 cols each)

  const int blockM = blockIdx.y * 128;
  const int blockN = blockIdx.x * 128;

  v8f acc[4][2];
  const v8f vzero = {0.f, 0.f, 0.f, 0.f, 0.f, 0.f, 0.f, 0.f};
#pragma unroll
  for (int i = 0; i < 4; ++i)
#pragma unroll
    for (int j = 0; j < 2; ++j) acc[i][j] = vzero;

  const int arow = tid >> 1;        // 0..127 : A tile row       (fallback path)
  const int ahalf = (tid & 1) * 16; // 0/16   : 32B half of row  (fallback path)
  const int brow = tid >> 3;        // 0..31  : B tile K-row
  const int bn0 = (tid & 7) * 16;   // 0..112 : 16-column segment
  (void)arow; (void)ahalf;

  const int nt = K >> 5;

  // ---- prologue: stage K-tile 0 into buffer 0 ----
#if USE_TDM
  if (wid == 0)
    tdm_load_tile_a(A + (ull)blockM * lda, (unsigned)(ull)&sA[0][0], lda);
#else
  {
    const u32x4* src = (const u32x4*)(A + (ull)(blockM + arow) * lda + ahalf);
    u32x4 d0 = src[0];
    u32x4 d1 = src[1];
    u32x4* dst = (u32x4*)&sA[0][arow * 40 + ahalf];
    dst[0] = d0;
    dst[1] = d1;
  }
#endif
  {
    const u32x4* src = (const u32x4*)(Bm + (ull)brow * ldb + blockN + bn0);
    union { u32x4 q[2]; bf16 h[16]; } t;
    t.q[0] = src[0];
    t.q[1] = src[1];
#pragma unroll
    for (int i = 0; i < 16; ++i) sB[0][(bn0 + i) * 40 + brow] = t.h[i];
  }
#if USE_TDM
  if (wid == 0) __builtin_amdgcn_s_wait_tensorcnt(0);
#endif
  __syncthreads();

  for (int kt = 0; kt < nt; ++kt) {
    const int cur = kt & 1;
    const int nxt = cur ^ 1;
    const bool pref = (kt + 1) < nt;

    // ---- kick off next tile: TDM for A, global loads for B (in-flight
    //      during the WMMA block below) ----
    u32x4 bp0 = {}, bp1 = {};
#if !USE_TDM
    u32x4 ap0 = {}, ap1 = {};
#endif
    if (pref) {
#if USE_TDM
      if (wid == 0)
        tdm_load_tile_a(A + (ull)blockM * lda + (kt + 1) * 32,
                        (unsigned)(ull)&sA[nxt][0], lda);
#else
      const u32x4* asrc =
          (const u32x4*)(A + (ull)(blockM + arow) * lda + (kt + 1) * 32 + ahalf);
      ap0 = asrc[0];
      ap1 = asrc[1];
#endif
      const u32x4* bsrc =
          (const u32x4*)(Bm + (ull)((kt + 1) * 32 + brow) * ldb + blockN + bn0);
      bp0 = bsrc[0];
      bp1 = bsrc[1];
    }

    // ---- compute from buffer `cur` ----
    // B fragments: lane = column n; K runs [kg*8, kg*8+8) then [16+kg*8, ...)
    // -> two 16B runs in the [n][k] staging layout.
    v16bf bfrag[2];
#pragma unroll
    for (int fn = 0; fn < 2; ++fn) {
      const int n = wn * 32 + fn * 16 + l16;
      union { u32x4 q[2]; v16bf v; } u;
      u.q[0] = *(const u32x4*)&sB[cur][n * 40 + kg * 8];
      u.q[1] = *(const u32x4*)&sB[cur][n * 40 + 16 + kg * 8];
      bfrag[fn] = u.v;
    }
    // A fragments: lane = row m; K runs [kg*8, kg*8+8) then [16+kg*8, ...).
#pragma unroll
    for (int fm = 0; fm < 4; ++fm) {
      const int m = wm * 64 + fm * 16 + l16;
      union { u32x4 q[2]; v16bf v; } u;
      u.q[0] = *(const u32x4*)&sA[cur][m * 40 + kg * 8];
      u.q[1] = *(const u32x4*)&sA[cur][m * 40 + 16 + kg * 8];
      const v16bf afrag = u.v;
#pragma unroll
      for (int fn = 0; fn < 2; ++fn) {
        acc[fm][fn] = __builtin_amdgcn_wmma_f32_16x16x32_bf16(
            false, afrag, false, bfrag[fn], (short)0, acc[fm][fn], false, false);
      }
    }

    // ---- finish staging the next tile into the other buffer ----
    if (pref) {
#if !USE_TDM
      u32x4* adst = (u32x4*)&sA[nxt][arow * 40 + ahalf];
      adst[0] = ap0;
      adst[1] = ap1;
#endif
      union { u32x4 q[2]; bf16 h[16]; } t;
      t.q[0] = bp0;
      t.q[1] = bp1;
#pragma unroll
      for (int i = 0; i < 16; ++i) sB[nxt][(bn0 + i) * 40 + brow] = t.h[i];
    }
#if USE_TDM
    if (wid == 0) __builtin_amdgcn_s_wait_tensorcnt(0);
#endif
    __syncthreads();
  }

  // ---- epilogue: C/D layout = lane n = l16, m = (lane>>4)*8 + j ----
  const int lhi = lane >> 4;
#pragma unroll
  for (int fm = 0; fm < 4; ++fm) {
#pragma unroll
    for (int fn = 0; fn < 2; ++fn) {
      const int col = blockN + wn * 32 + fn * 16 + l16;
      const float bval = HAS_BIAS ? bias[col] : 0.0f;
#pragma unroll
      for (int j = 0; j < 8; ++j) {
        const int row = blockM + wm * 64 + fm * 16 + lhi * 8 + j;
        const float v = acc[fm][fn][j] * alpha + bval;
        if (OUT_MODE == 0) {
          ((float*)Cv)[coff + (ull)row * ldc + col] = v;
        } else if (OUT_MODE == 1) {
          ((bf16*)Cv)[coff + (ull)row * ldc + col] = (bf16)v;
        } else {
          const int rhi = row >> 10, rlo = row & 1023;
          ((bf16*)Cv)[coff + (ull)rhi * strideRB + (ull)col * ldtrans + rlo] =
              (bf16)v;
        }
      }
    }
  }
}

// ---------------------------------------------------------------------------
// Pipeline (linear-attention reassociation, no softmax in reference):
//  P0  convert q,k,v,Wq,Wk,Wv,Wo  fp32 -> bf16
//  P1  qh = q@Wq[h]+bq   vh = v@Wv[h]+bv   khT = (k@Wk[h]+bk)^T per (h,b)
//  P2  KtV[h,b] = khT[h,b] @ vh[h,b]
//  P3  cat[b,s,h*E+e] = (qh[h,b] @ KtV[h,b]) / E    (== scores@vh/E)
//  P4  out = cat @ Wo + bo  (fp32 output)
// ---------------------------------------------------------------------------
extern "C" void kernel_launch(void* const* d_in, const int* in_sizes, int n_in,
                              void* d_out, int out_size, void* d_ws, size_t ws_size,
                              hipStream_t stream) {
  (void)in_sizes; (void)n_in; (void)out_size; (void)ws_size;
  constexpr int Bb = 8, S = 1024, E = 512, H = 8;
  constexpr int M = Bb * S;   // 8192
  constexpr int HE = H * E;   // 4096

  const float* bq = (const float*)d_in[4];
  const float* bk = (const float*)d_in[6];
  const float* bv = (const float*)d_in[8];
  const float* bo = (const float*)d_in[10];

  char* w = (char*)d_ws;
  const size_t MBy = 1ull << 20;
  bf16* q_bf  = (bf16*)(w + 0 * MBy);    //  8 MiB
  bf16* k_bf  = (bf16*)(w + 8 * MBy);    //  8 MiB
  bf16* v_bf  = (bf16*)(w + 16 * MBy);   //  8 MiB
  bf16* Wq_bf = (bf16*)(w + 24 * MBy);   //  4 MiB
  bf16* Wk_bf = (bf16*)(w + 28 * MBy);   //  4 MiB
  bf16* Wv_bf = (bf16*)(w + 32 * MBy);   //  4 MiB
  bf16* qh    = (bf16*)(w + 36 * MBy);   // 64 MiB  [H][B*S][E]
  bf16* khT   = (bf16*)(w + 100 * MBy);  // 64 MiB  [H][B][E][S]
  bf16* vh    = (bf16*)(w + 164 * MBy);  // 64 MiB  [H][B*S][E]
  bf16* ktv   = (bf16*)(w + 228 * MBy);  // 32 MiB  [H*B][E][E]
  bf16* Wo_bf = (bf16*)(w + 260 * MBy);  //  4 MiB
  bf16* cat   = (bf16*)(w + 100 * MBy);  // 64 MiB, reuses khT/vh after P2

  auto cvt = [&](int idx, bf16* dst, int n) {
    f32_to_bf16_kernel<<<dim3((n + 255) / 256), dim3(256), 0, stream>>>(
        (const float*)d_in[idx], dst, n);
  };
  cvt(0, q_bf, M * E);
  cvt(1, k_bf, M * E);
  cvt(2, v_bf, M * E);
  cvt(3, Wq_bf, H * E * E);
  cvt(5, Wk_bf, H * E * E);
  cvt(7, Wv_bf, H * E * E);
  cvt(9, Wo_bf, HE * E);

  dim3 blk(256);

  // ---- Pass 1: projections (z = h) ----
  dim3 g1(E / 128, M / 128, H);
  gemm_bf16_wmma<1, true><<<g1, blk, 0, stream>>>(
      q_bf, Wq_bf, qh, bq, E, E, E, E, 1,
      0, 0, (ull)E * E, 0, (ull)M * E, 0, (ull)E, 0, 0, 1.0f);
  gemm_bf16_wmma<2, true><<<g1, blk, 0, stream>>>(
      k_bf, Wk_bf, khT, bk, E, E, E, 0, 1,
      0, 0, (ull)E * E, 0, (ull)M * E, 0, (ull)E, (ull)E * S, S, 1.0f);
  gemm_bf16_wmma<1, true><<<g1, blk, 0, stream>>>(
      v_bf, Wv_bf, vh, bv, E, E, E, E, 1,
      0, 0, (ull)E * E, 0, (ull)M * E, 0, (ull)E, 0, 0, 1.0f);

  // ---- Pass 2: KtV[h,b] = khT[h,b] @ vh[h,b]  (M=N=512, K=1024; z = h*B+b) ----
  dim3 g2(E / 128, E / 128, H * Bb);
  gemm_bf16_wmma<1, false><<<g2, blk, 0, stream>>>(
      khT, vh, ktv, nullptr, S, S, E, E, 1,
      (ull)E * S, 0, (ull)S * E, 0, (ull)E * E, 0, 0, 0, 0, 1.0f);

  // ---- Pass 3: cat = (qh @ KtV)/E scattered into [B][S][H*E]  (z1=h, z2=b) ----
  dim3 g3(E / 128, S / 128, H * Bb);
  gemm_bf16_wmma<1, false><<<g3, blk, 0, stream>>>(
      qh, ktv, cat, nullptr, E, E, E, HE, Bb,
      (ull)M * E, (ull)S * E,
      (ull)Bb * E * E, (ull)E * E,
      (ull)E, (ull)S * HE,
      0, 0, 0, 1.0f / (float)E);

  // ---- Pass 4: out = cat @ Wo + bo  (fp32 out, M=8192, K=4096, N=512) ----
  dim3 g4(E / 128, M / 128, 1);
  gemm_bf16_wmma<0, true><<<g4, blk, 0, stream>>>(
      cat, Wo_bf, d_out, bo, HE, HE, E, E, 1,
      0, 0, 0, 0, 0, 0, 0, 0, 0, 1.0f);
}
